// Router_69604239999272
// MI455X (gfx1250) — matrix operand order repaired and checked by reading, not compiled, not run on hardware
//
#include <hip/hip_runtime.h>
#include <hip/hip_bf16.h>

// ---------------------------------------------------------------------------
// Fused MoE router for MI455X (gfx1250, wave32, WMMA + async global->LDS)
//   h      = tanh(X @ W1^T)            [8192 x 128]   <- bf16 WMMA, f32 accum
//   logits = h @ W2^T + PRR @ PGW^T    [8192 x 16]    <- VALU (tiny)
//   softmax -> top-2 -> mask expert15 -> renorm -> combine scatter
// X streamed from HBM exactly once (~134 MB => ~6us roofline floor @23.3TB/s).
// Per K-chunk: async HBM->LDS (ASYNCcnt) -> WMMA pipeline (sched_group_barrier
// enforced: 12 DS reads, then 12x[1 WMMA, 2 DS], then 4 WMMA) -> cvt bf16.
// ---------------------------------------------------------------------------

#define T_TOK  8192
#define DIM    4096
#define NEXP   16
#define HID    128          // 8 * NEXP
#define MTILE  64           // tokens per workgroup
#define KC     64           // K-chunk staged per iteration
#define NKC    (DIM / KC)   // 64 iterations
#define APITCH 72           // ushorts per bf16 LDS row: 64 + 8 pad (144 B)
#define RPITCH 68           // floats per raw f32 LDS row: 64 + 4 pad (272 B)
#define HPITCH 129          // floats per h row (pad 1 -> conflict-free)

typedef __attribute__((ext_vector_type(16))) __bf16 v16bf;
typedef __attribute__((ext_vector_type(8)))  __bf16 v8bf;
typedef __attribute__((ext_vector_type(8)))  float  v8f;
typedef __attribute__((ext_vector_type(4)))  int    v4i;

union Frag { v16bf v; v8bf h[2]; };

#define HAS_ASYNC_LDS __has_builtin(__builtin_amdgcn_global_load_async_to_lds_b128)

typedef __attribute__((address_space(1))) v4i* gptr_v4i;   // global int4*
typedef __attribute__((address_space(3))) v4i* lptr_v4i;   // LDS int4*

__device__ __forceinline__ void async_cp16(const float* gp, float* lp) {
#if HAS_ASYNC_LDS
    __builtin_amdgcn_global_load_async_to_lds_b128(
        (gptr_v4i)(gp), (lptr_v4i)(lp), 0, 0);
#else
    *(float4*)lp = *(const float4*)gp;
#endif
}

__device__ __forceinline__ void wait_async0() {
#if __has_builtin(__builtin_amdgcn_s_wait_asynccnt)
    __builtin_amdgcn_s_wait_asynccnt(0);
#else
    asm volatile("s_wait_asynccnt 0x0" ::: "memory");
#endif
}

// Explicit instruction-scheduling controls (no-ops if builtins absent)
#if __has_builtin(__builtin_amdgcn_sched_barrier)
#define SCHED_FENCE() __builtin_amdgcn_sched_barrier(0)
#else
#define SCHED_FENCE()
#endif
#if __has_builtin(__builtin_amdgcn_sched_group_barrier)
#define SCHED_GROUP(mask, size, id) __builtin_amdgcn_sched_group_barrier(mask, size, id)
#else
#define SCHED_GROUP(mask, size, id)
#endif
// mask bits: 0x008 = MFMA/WMMA, 0x100 = DS read

// f32x4 -> packed bf16x4 (RNE); clang lowers to v_cvt_pk_bf16_f32 pairs
__device__ __forceinline__ unsigned long long pk4(float4 f) {
    union { __bf16 b[4]; unsigned long long u; } r;
    r.b[0] = (__bf16)f.x; r.b[1] = (__bf16)f.y;
    r.b[2] = (__bf16)f.z; r.b[3] = (__bf16)f.w;
    return r.u;
}

__global__ __launch_bounds__(128)
__attribute__((amdgpu_waves_per_eu(2)))
void moe_router_fused(const float* __restrict__ X,
                      const float* __restrict__ PRR,
                      const float* __restrict__ W1,
                      const float* __restrict__ W2,
                      const float* __restrict__ PGW,
                      float* __restrict__ out) {
    // LDS ~146 KB: raw 52.2K + bf16 tiles 55.3K + H 33K + W2/PGW 9.2K
    __shared__ __align__(16) float Araw[MTILE * RPITCH];            // 17408 B
    __shared__ __align__(16) float Braw[HID * RPITCH];              // 34816 B
    __shared__ __align__(16) unsigned short Abuf[2][MTILE * APITCH];// 18432 B
    __shared__ __align__(16) unsigned short Bbuf[2][HID * APITCH];  // 36864 B
    __shared__ __align__(16) float Hbuf[MTILE * HPITCH];            // 33024 B
    __shared__ float W2s[NEXP * HID];
    __shared__ float PGWs[NEXP * NEXP];

    const int tid  = threadIdx.x;
    const int M0   = blockIdx.x * MTILE;
    const int lane = tid & 31;
    const int wav  = tid >> 5;       // 4 waves: wave w owns token rows [w*16, w*16+16)
    const int lr   = lane & 15;
    const int hf   = lane >> 4;      // lane half (ISA 16-bit operand layout)

    // Preload the tiny GEMM2 weights once (L2 -> LDS)
    for (int i = tid; i < NEXP * HID;  i += 128) W2s[i]  = W2[i];
    for (int i = tid; i < NEXP * NEXP; i += 128) PGWs[i] = PGW[i];

    v8f acc[8];
    #pragma unroll
    for (int n = 0; n < 8; ++n)
        #pragma unroll
        for (int r = 0; r < 8; ++r) acc[n][r] = 0.0f;

    // Phase (a): async-copy raw f32 chunk HBM->LDS (no VGPR staging, ASYNCcnt).
    // Wave w copies A rows [w*16,w*16+16) and B rows [w*32,w*32+32) so the
    // later per-wave convert only touches data its own asyncs wrote.
    auto stage_async = [&](int k0) {
        if (k0 + KC < DIM)   // distance-2 stream hint (global_prefetch_b8)
            __builtin_prefetch(X + (size_t)(M0 + (tid & 63)) * DIM + k0 + KC, 0, 3);
        #pragma unroll
        for (int i = 0; i < 8; ++i) {            // A: 16 rows x 16 float4 / wave
            int idx = lane * 8 + i;
            int row = wav * 16 + (idx >> 4), c4 = idx & 15;
            async_cp16(X + (size_t)(M0 + row) * DIM + k0 + c4 * 4,
                       &Araw[row * RPITCH + c4 * 4]);
        }
        #pragma unroll
        for (int i = 0; i < 16; ++i) {           // B: 32 rows x 16 float4 / wave
            int idx = lane * 16 + i;
            int row = wav * 32 + (idx >> 4), c4 = idx & 15;
            async_cp16(W1 + (size_t)row * DIM + k0 + c4 * 4,
                       &Braw[row * RPITCH + c4 * 4]);
        }
    };

    // Phase (c): raw f32 -> bf16 tiles (K-contiguous rows are exactly the
    // byte order the 16-bit WMMA A/B fragments want; padded pitches).
    auto convert = [&](int buf) {
        #pragma unroll
        for (int i = 0; i < 8; ++i) {
            int idx = lane * 8 + i;
            int row = wav * 16 + (idx >> 4), c4 = idx & 15;
            float4 f = *(const float4*)&Araw[row * RPITCH + c4 * 4];
            *(unsigned long long*)&Abuf[buf][row * APITCH + c4 * 4] = pk4(f);
        }
        #pragma unroll
        for (int i = 0; i < 16; ++i) {
            int idx = lane * 16 + i;
            int row = wav * 32 + (idx >> 4), c4 = idx & 15;
            float4 f = *(const float4*)&Braw[row * RPITCH + c4 * 4];
            *(unsigned long long*)&Bbuf[buf][row * APITCH + c4 * 4] = pk4(f);
        }
    };

    // B fragment (32x16): lane = column N, K = 8*hf+[0,8) and 16+8*hf+[0,8)
    auto loadB = [&](int cur, int s) -> Frag {
        const int k32 = s >> 3, n = s & 7;
        const unsigned short* bb =
            &Bbuf[cur][(n * 16 + lr) * APITCH + k32 * 32 + hf * 8];
        Frag b;
        b.h[0] = *(const v8bf*)(bb);
        b.h[1] = *(const v8bf*)(bb + 16);
        return b;
    };

    // Phase (b): 36 DS reads + 16 WMMAs, fenced into their own scheduling
    // region; sched_group_barrier forces the pipeline
    //   [12 DS] then 12x[1 WMMA, 2 DS] then [4 WMMA]
    // so each WMMA consumes a fragment loaded ~3 WMMAs earlier.
    auto compute = [&](int cur) {
        SCHED_FENCE();
        const unsigned short* ab =
            &Abuf[cur][(wav * 16 + lr) * APITCH + hf * 8];
        Frag a0, a1;                 // A fragments for both K32 steps
        a0.h[0] = *(const v8bf*)(ab);
        a0.h[1] = *(const v8bf*)(ab + 16);
        a1.h[0] = *(const v8bf*)(ab + 32);
        a1.h[1] = *(const v8bf*)(ab + 48);
        #pragma unroll
        for (int s = 0; s < 16; ++s) {
            Frag b = loadB(cur, s);
            const int n = s & 7;
            acc[n] = __builtin_amdgcn_wmma_f32_16x16x32_bf16(
                false, ((s >> 3) ? a1 : a0).v, false, b.v,
                (short)0, acc[n], false, false);
        }
        SCHED_GROUP(0x100, 12, 0);            // a0,a1,b0..b3 loads up front
        #pragma unroll
        for (int s = 0; s < 12; ++s) {
            SCHED_GROUP(0x008, 1, 0);         // WMMA_s
            SCHED_GROUP(0x100, 2, 0);         // loads for b_{s+4}
        }
        SCHED_GROUP(0x008, 4, 0);             // drain: last 4 WMMAs
        SCHED_FENCE();
    };

    stage_async(0);
    wait_async0();
    convert(0);
    __syncthreads();

    for (int kc = 0; kc < NKC; ++kc) {
        const int cur = kc & 1;
        const bool more = (kc + 1 < NKC);
        if (more) stage_async((kc + 1) * KC);  // HBM->LDS under compute
        compute(cur);                          // only frag ds_loads use DScnt
        if (more) { wait_async0(); convert(cur ^ 1); }
        __syncthreads();
    }

    // tanh + park h in LDS (C-layout: VGPR r, lanes<16 -> M=r, lanes>=16 -> M=r+8)
    #pragma unroll
    for (int n = 0; n < 8; ++n)
        #pragma unroll
        for (int r = 0; r < 8; ++r)
            Hbuf[(wav * 16 + hf * 8 + r) * HPITCH + n * 16 + lr] = tanhf(acc[n][r]);
    __syncthreads();

    // Epilogue: one token per thread (64 tokens / WG)
    if (tid < MTILE) {
        const int t = M0 + tid;
        float logits[NEXP];
        #pragma unroll
        for (int e = 0; e < NEXP; ++e) logits[e] = 0.0f;
        #pragma unroll
        for (int i = 0; i < NEXP; ++i) {
            float p = PRR[(size_t)t * NEXP + i];
            #pragma unroll
            for (int e = 0; e < NEXP; ++e) logits[e] += p * PGWs[e * NEXP + i];
        }
        for (int j = 0; j < HID; ++j) {
            float hv = Hbuf[tid * HPITCH + j];
            #pragma unroll
            for (int e = 0; e < NEXP; ++e) logits[e] += hv * W2s[e * HID + j];
        }
        // softmax over 16
        float m = logits[0];
        #pragma unroll
        for (int e = 1; e < NEXP; ++e) m = fmaxf(m, logits[e]);
        float ex[NEXP], s = 0.0f;
        #pragma unroll
        for (int e = 0; e < NEXP; ++e) { ex[e] = __expf(logits[e] - m); s += ex[e]; }
        const float inv = 1.0f / s;
        // top-2 (strict > => lowest index wins ties, matching jax.lax.top_k)
        int i1 = 0; float b1 = ex[0] * inv;
        #pragma unroll
        for (int e = 1; e < NEXP; ++e) {
            float w = ex[e] * inv;
            if (w > b1) { b1 = w; i1 = e; }
        }
        int i2 = (i1 == 0) ? 1 : 0; float b2 = ex[i2] * inv;
        #pragma unroll
        for (int e = 0; e < NEXP; ++e) {
            if (e == i1 || e == i2) continue;
            float w = ex[e] * inv;
            if (w > b2) { b2 = w; i2 = e; }
        }
        // mask routes to last expert, renormalize over the 2 slots
        float s1 = (i1 == NEXP - 1) ? 0.0f : b1;
        float s2 = (i2 == NEXP - 1) ? 0.0f : b2;
        const float rs = s1 + s2;
        s1 /= rs; s2 /= rs;

        float* out_logits = out;
        float* out_selw   = out + (size_t)T_TOK * NEXP;
        int*   out_seli   = (int*)(out + (size_t)T_TOK * (NEXP + 2));
        float* out_comb   = out + (size_t)T_TOK * (NEXP + 4);

        #pragma unroll
        for (int e = 0; e < NEXP; ++e) out_logits[(size_t)t * NEXP + e] = logits[e];
        out_selw[t * 2 + 0] = s1;
        out_selw[t * 2 + 1] = s2;
        out_seli[t * 2 + 0] = i1;
        out_seli[t * 2 + 1] = i2;
        float comb[NEXP];
        #pragma unroll
        for (int e = 0; e < NEXP; ++e) comb[e] = 0.0f;
        comb[i1] = s1;
        comb[i2] = s2;
        #pragma unroll
        for (int e = 0; e < NEXP; ++e) out_comb[(size_t)t * NEXP + e] = comb[e];
    }
}

extern "C" void kernel_launch(void* const* d_in, const int* in_sizes, int n_in,
                              void* d_out, int out_size, void* d_ws, size_t ws_size,
                              hipStream_t stream) {
    const float* X   = (const float*)d_in[0];   // [8192, 4096]
    const float* PRR = (const float*)d_in[1];   // [8192, 16]
    const float* W1  = (const float*)d_in[2];   // [128, 4096]
    const float* W2  = (const float*)d_in[3];   // [16, 128]
    const float* PGW = (const float*)d_in[4];   // [16, 16]
    float* out = (float*)d_out;                 // logits | sel_w | sel_idx(i32) | combine

    dim3 grid(T_TOK / MTILE);                   // 128 workgroups
    dim3 block(128);                            // 4 wave32s
    hipLaunchKernelGGL(moe_router_fused, grid, block, 0, stream,
                       X, PRR, W1, W2, PGW, out);
}
